// NeuroSAT_39934605918418
// MI455X (gfx1250) — compile-verified, matrix operand order checked by reading, not run on hardware
//
#include <hip/hip_runtime.h>
#include <hip/hip_bf16.h>

// ---------------------------------------------------------------------------
// NeuroSAT on gfx1250 (MI455X):
//  - embeddings in bf16; GEMMs on v_wmma_f32_16x16x32_bf16
//  - 32-row M-tiles per wave: each B fragment (LDS) feeds 2 WMMAs
//  - weights LDS-resident; async global<->LDS staging, offset-based dbl-buffer
//  - COO segment-sum: f32 global atomics + cvt pass; scales folded into W1
// ---------------------------------------------------------------------------

#define NUM_VARS    100000
#define NUM_LITS    200000
#define NUM_CLAUSES 420000
#define NNZ         (3 * NUM_CLAUSES)
#define DD          128
#define RR          4

typedef __attribute__((ext_vector_type(16))) __bf16         v16bf;
typedef __attribute__((ext_vector_type(8)))  float          v8f;
typedef __attribute__((ext_vector_type(16))) unsigned short v16us;
typedef int v4i __attribute__((vector_size(16)));

#define AS1 __attribute__((address_space(1)))
#define AS3 __attribute__((address_space(3)))

#define HAS_ASYNC_LD  __has_builtin(__builtin_amdgcn_global_load_async_to_lds_b128)
#define HAS_ASYNC_ST  __has_builtin(__builtin_amdgcn_global_store_async_from_lds_b128)
#define HAS_WAIT_ASY  __has_builtin(__builtin_amdgcn_s_wait_asynccnt)
#define HAS_WAIT_DS   __has_builtin(__builtin_amdgcn_s_wait_dscnt)

__device__ __forceinline__ unsigned short f2bf_bits(float f) {
  return __builtin_bit_cast(unsigned short, (__bf16)f);   // native v_cvt path
}
__device__ __forceinline__ float bf_bits2f(unsigned short b) {
  return __builtin_bit_cast(float, ((unsigned)b) << 16);
}

// --- async copy helpers -----------------------------------------------------
__device__ __forceinline__ void cp_g2l_b128(unsigned short* lds, const unsigned short* g) {
#if HAS_ASYNC_LD
  __builtin_amdgcn_global_load_async_to_lds_b128(
      (AS1 v4i*)(unsigned short*)g, (AS3 v4i*)lds, 0, 0);
#else
  *(v16us*)lds = *(const v16us*)g;
#endif
}
__device__ __forceinline__ void cp_l2g_b128(unsigned short* g, const unsigned short* lds) {
#if HAS_ASYNC_ST
  __builtin_amdgcn_global_store_async_from_lds_b128(
      (AS1 v4i*)g, (AS3 v4i*)(unsigned short*)lds, 0, 0);
#else
  *(v16us*)g = *(const v16us*)lds;
#endif
}
__device__ __forceinline__ void wait_async0() {
#if HAS_ASYNC_LD || HAS_ASYNC_ST
#if HAS_WAIT_ASY
  __builtin_amdgcn_s_wait_asynccnt(0);
#else
  asm volatile("s_wait_asynccnt 0" ::: "memory");
#endif
#endif
}
__device__ __forceinline__ void wait_ds0() {
#if HAS_WAIT_DS
  __builtin_amdgcn_s_wait_dscnt(0);
#else
  asm volatile("s_wait_dscnt 0" ::: "memory");
#endif
}

// --- fragment helpers -------------------------------------------------------
__device__ __forceinline__ v16bf make_afrag_lds(const unsigned short* p, int hi) {
  p += hi * 8;
  v16bf a;
#pragma unroll
  for (int i = 0; i < 8; ++i) a[i]     = __builtin_bit_cast(__bf16, p[i]);
#pragma unroll
  for (int i = 0; i < 8; ++i) a[8 + i] = __builtin_bit_cast(__bf16, p[16 + i]);
  return a;
}
__device__ __forceinline__ v16bf load_bfrag(const unsigned short* p, int lane) {
  v16us t = *(const v16us*)(p + lane * 16);
  return __builtin_bit_cast(v16bf, t);
}

#define WMMA_BF16(a, b, c) \
  __builtin_amdgcn_wmma_f32_16x16x32_bf16(false, (a), false, (b), (short)0, (c), false, false)

// One MLP layer on a 32-row M-tile: every B fragment is reused for 2 WMMAs.
template <int KSTEPS>
__device__ __forceinline__ void mlp_layer32(const unsigned short* x_lds, int xstride,
                                            const unsigned short* w_lds, int lane,
                                            v8f acc0[8], v8f acc1[8]) {
  const int hi = lane >> 4, cb = lane & 15;
  const unsigned short* xr0 = x_lds + cb * xstride;
  const unsigned short* xr1 = x_lds + (16 + cb) * xstride;
#pragma unroll
  for (int ks = 0; ks < KSTEPS; ++ks) {
    v16bf a0 = make_afrag_lds(xr0 + ks * 32, hi);
    v16bf a1 = make_afrag_lds(xr1 + ks * 32, hi);
    const unsigned short* wp = w_lds + ks * (8 * 512);
#pragma unroll
    for (int n = 0; n < 8; ++n) {
      v16bf b = load_bfrag(wp + n * 512, lane);
      acc0[n] = WMMA_BF16(a0, b, acc0[n]);
      acc1[n] = WMMA_BF16(a1, b, acc1[n]);
    }
  }
}

// bias (+optional relu) -> bf16 16x128 half-tile in LDS (wave-private)
__device__ __forceinline__ void store_h(const v8f acc[8], const float* b,
                                        unsigned short* h, int lane, bool relu) {
  const int hi = lane >> 4, cb = lane & 15;
#pragma unroll
  for (int n = 0; n < 8; ++n) {
    const int col = n * 16 + cb;
    const float bv = b[col];
#pragma unroll
    for (int r = 0; r < 8; ++r) {
      float v = acc[n][r] + bv;
      if (relu) v = v > 0.f ? v : 0.f;
      h[(hi * 8 + r) * DD + col] = f2bf_bits(v);
    }
  }
}

// stage one source: 32 rows x 128 bf16 into the X tile at column offset colo.
__device__ __forceinline__ void stage_src32(const unsigned short* g, unsigned short* x,
                                            int xstride, int colo, int lane) {
#pragma unroll
  for (int i = 0; i < 16; ++i) {
    int idx = i * 32 + lane;                 // 16B chunk id (512 chunks)
    int r = idx >> 4, c = (idx & 15) * 8;    // 8 u16 per chunk
    cp_g2l_b128(x + r * xstride + colo + c, g + (size_t)r * DD + c);
  }
}
// block-cooperative weight staging into LDS
__device__ __forceinline__ void stage_g2l(unsigned short* l, const unsigned short* g,
                                          int n, int tid, int nthreads) {
  for (int i = tid * 8; i < n; i += nthreads * 8) cp_g2l_b128(l + i, g + i);
}

// ---------------------------------------------------------------------------
// Weight packer: f32 [K x 128] row-major -> bf16 fragment order
// [kstep][ntile][lane][16]; rows in [ksc0,ksc1) get multiplied by *scale_p.
// ---------------------------------------------------------------------------
__global__ void pack_w_kernel(const float* __restrict__ W, unsigned short* __restrict__ Wp,
                              int ksteps, const float* __restrict__ scale_p,
                              int ksc0, int ksc1) {
  int idx = blockIdx.x * blockDim.x + threadIdx.x;
  int total = ksteps * 8 * 32;
  if (idx >= total) return;
  const float s = scale_p[0];
  int lane = idx & 31;
  int nt   = (idx >> 5) & 7;
  int ks   = idx >> 8;
  int col  = nt * 16 + (lane & 15);
  int kb   = ks * 32 + (lane >> 4) * 8;
  unsigned short* dst = Wp + (size_t)idx * 16;
#pragma unroll
  for (int i = 0; i < 8; ++i) {
    int k = kb + i;
    float sv = (k >= ksc0 && k < ksc1) ? s : 1.0f;
    dst[i] = f2bf_bits(W[(size_t)k * DD + col] * sv);
  }
#pragma unroll
  for (int i = 0; i < 8; ++i) {
    int k = kb + 16 + i;
    float sv = (k >= ksc0 && k < ksc1) ? s : 1.0f;
    dst[8 + i] = f2bf_bits(W[(size_t)k * DD + col] * sv);
  }
}

__global__ void init_embed_kernel(unsigned short* __restrict__ p,
                                  const float* __restrict__ scale_p, size_t n) {
  size_t i = (size_t)blockIdx.x * blockDim.x + threadIdx.x;
  unsigned short v = f2bf_bits(scale_p[0]);
  if (i < n) p[i] = v;
}

// f32 accumulator -> bf16 operand buffer (4 elems/thread, packed 8B store)
__global__ void cvt_kernel(const float* __restrict__ s, unsigned short* __restrict__ d,
                           size_t n4) {
  size_t i = (size_t)blockIdx.x * blockDim.x + threadIdx.x;
  if (i >= n4) return;
  float4 v = ((const float4*)s)[i];
  uint2 r;
  r.x = (unsigned)f2bf_bits(v.x) | ((unsigned)f2bf_bits(v.y) << 16);
  r.y = (unsigned)f2bf_bits(v.z) | ((unsigned)f2bf_bits(v.w) << 16);
  *(uint2*)(d + i * 4) = r;
}

// ---------------------------------------------------------------------------
// COO segment-sum: one wave per edge; bf16 row gather, f32 global atomics
// ---------------------------------------------------------------------------
__global__ void __launch_bounds__(256) scatter_kernel(const unsigned short* __restrict__ src,
                                                      const int* __restrict__ seg_out,
                                                      const int* __restrict__ seg_in,
                                                      float* __restrict__ dst) {
  int w = (blockIdx.x * blockDim.x + threadIdx.x) >> 5;
  int lane = threadIdx.x & 31;
  if (w >= NNZ) return;
  int o = seg_out[w];
  int i = seg_in[w];
  uint2 q = *(const uint2*)(src + (size_t)i * DD + lane * 4);
  float* d = dst + (size_t)o * DD + lane * 4;
  atomicAdd(d + 0, bf_bits2f((unsigned short)(q.x & 0xffff)));
  atomicAdd(d + 1, bf_bits2f((unsigned short)(q.x >> 16)));
  atomicAdd(d + 2, bf_bits2f((unsigned short)(q.y & 0xffff)));
  atomicAdd(d + 3, bf_bits2f((unsigned short)(q.y >> 16)));
}

// ---------------------------------------------------------------------------
// Clause MLP: C = relu([C | LC] @ W1s + b1) @ W2 + b2    (bf16, 32-row tiles)
// LDS: 96KB weights + 4 waves x (h 8KB + X dbl-buf 32KB) = 256KB
// ---------------------------------------------------------------------------
__global__ void __launch_bounds__(128) clause_mlp_kernel(
    const unsigned short* Cin, const unsigned short* LC,
    const unsigned short* W1p, const float* __restrict__ b1,
    const unsigned short* W2p, const float* __restrict__ b2,
    unsigned short* Cout) {
  extern __shared__ unsigned short smem[];
  const int XS = 256, WVS = 4;
  unsigned short* w1 = smem;                       // 8*8*512  = 32768 u16
  unsigned short* w2 = w1 + 32768;                 // 4*8*512  = 16384 u16
  const int wave = threadIdx.x >> 5, lane = threadIdx.x & 31;
  unsigned short* wb    = w2 + 16384 + wave * (4096 + 2 * 32 * XS);
  unsigned short* h     = wb;                      // 32 x 128
  unsigned short* xbase = wb + 4096;               // two 32*XS buffers

  stage_g2l(w1, W1p, 32768, threadIdx.x, 32 * WVS);
  stage_g2l(w2, W2p, 16384, threadIdx.x, 32 * WVS);
  wait_async0();
  __syncthreads();

  const int ntiles = NUM_CLAUSES / 32;
  const int stride = gridDim.x * WVS;
  int t = blockIdx.x * WVS + wave;
  if (t < ntiles) {
    stage_src32(Cin + (size_t)t * 32 * DD, xbase, XS, 0, lane);
    stage_src32(LC  + (size_t)t * 32 * DD, xbase, XS, DD, lane);
  }
  wait_async0();
  int cur = 0;
  for (; t < ntiles; t += stride) {
    unsigned short* xc = xbase + cur * (32 * XS);        // offset-based select
    unsigned short* xn = xbase + (cur ^ 1) * (32 * XS);  // (keeps AS3 inference)
    int tn = t + stride;
    if (tn < ntiles) {
      stage_src32(Cin + (size_t)tn * 32 * DD, xn, XS, 0, lane);
      stage_src32(LC  + (size_t)tn * 32 * DD, xn, XS, DD, lane);
    }
    v8f acc0[8] = {}, acc1[8] = {};
    mlp_layer32<8>(xc, XS, w1, lane, acc0, acc1);
    store_h(acc0, b1, h, lane, true);
    store_h(acc1, b1, h + 16 * DD, lane, true);
    v8f o0[8] = {}, o1[8] = {};
    mlp_layer32<4>(h, DD, w2, lane, o0, o1);
    store_h(o0, b2, h, lane, false);
    store_h(o1, b2, h + 16 * DD, lane, false);
    wait_ds0();                                  // h visible to async engine
    unsigned short* og = Cout + (size_t)t * 32 * DD;
#pragma unroll
    for (int i = 0; i < 16; ++i) {
      int idx = i * 32 + lane;
      int r = idx >> 4, c = (idx & 15) * 8;
      cp_l2g_b128(og + r * DD + c, h + r * DD + c);
    }
    wait_async0();                               // next X buf + store done
    cur ^= 1;
  }
}

// ---------------------------------------------------------------------------
// Literal MLP: Lout = relu([L | CL | flipL] @ W1s + b1) @ W2 + b2
// LDS: 128KB weights + 3 waves x (h 8KB + X dbl-buf 48KB) = 296KB -> 96 thr
// ---------------------------------------------------------------------------
__global__ void __launch_bounds__(96) lit_mlp_kernel(
    const unsigned short* __restrict__ Lin, const unsigned short* __restrict__ CL,
    const unsigned short* W1p, const float* __restrict__ b1,
    const unsigned short* W2p, const float* __restrict__ b2,
    unsigned short* __restrict__ Lout) {
  extern __shared__ unsigned short smem[];
  const int XS = 384, WVS = 3;
  unsigned short* w1 = smem;                       // 12*8*512 = 49152 u16
  unsigned short* w2 = w1 + 49152;                 // 16384 u16
  const int wave = threadIdx.x >> 5, lane = threadIdx.x & 31;
  unsigned short* wb    = w2 + 16384 + wave * (4096 + 2 * 32 * XS);
  unsigned short* h     = wb;
  unsigned short* xbase = wb + 4096;

  stage_g2l(w1, W1p, 49152, threadIdx.x, 32 * WVS);
  stage_g2l(w2, W2p, 16384, threadIdx.x, 32 * WVS);
  wait_async0();
  __syncthreads();

  const int ntiles = NUM_LITS / 32;
  const int stride = gridDim.x * WVS;
  int t = blockIdx.x * WVS + wave;
  if (t < ntiles) {
    int row0 = t * 32;
    int frow0 = (row0 < NUM_VARS) ? (row0 + NUM_VARS) : (row0 - NUM_VARS);
    stage_src32(Lin + (size_t)row0 * DD, xbase, XS, 0, lane);
    stage_src32(CL  + (size_t)row0 * DD, xbase, XS, DD, lane);
    stage_src32(Lin + (size_t)frow0 * DD, xbase, XS, 2 * DD, lane);
  }
  wait_async0();
  int cur = 0;
  for (; t < ntiles; t += stride) {
    unsigned short* xc = xbase + cur * (32 * XS);
    unsigned short* xn = xbase + (cur ^ 1) * (32 * XS);
    int tn = t + stride;
    if (tn < ntiles) {
      int row0 = tn * 32;
      int frow0 = (row0 < NUM_VARS) ? (row0 + NUM_VARS) : (row0 - NUM_VARS);
      stage_src32(Lin + (size_t)row0 * DD, xn, XS, 0, lane);
      stage_src32(CL  + (size_t)row0 * DD, xn, XS, DD, lane);
      stage_src32(Lin + (size_t)frow0 * DD, xn, XS, 2 * DD, lane);
    }
    v8f acc0[8] = {}, acc1[8] = {};
    mlp_layer32<12>(xc, XS, w1, lane, acc0, acc1);
    store_h(acc0, b1, h, lane, true);
    store_h(acc1, b1, h + 16 * DD, lane, true);
    v8f o0[8] = {}, o1[8] = {};
    mlp_layer32<4>(h, DD, w2, lane, o0, o1);
    store_h(o0, b2, h, lane, false);
    store_h(o1, b2, h + 16 * DD, lane, false);
    wait_ds0();
    unsigned short* og = Lout + (size_t)t * 32 * DD;
#pragma unroll
    for (int i = 0; i < 16; ++i) {
      int idx = i * 32 + lane;
      int r = idx >> 4, c = (idx & 15) * 8;
      cp_l2g_b128(og + r * DD + c, h + r * DD + c);
    }
    wait_async0();
    cur ^= 1;
  }
}

// ---------------------------------------------------------------------------
// Var head: 3x (GEMM+relu) + dot with v_W4 + b4 -> out[var] (f32)
// LDS: 128KB weights + 4 waves x 40KB = 288KB
// ---------------------------------------------------------------------------
__global__ void __launch_bounds__(128) var_mlp_kernel(
    const unsigned short* __restrict__ L,
    const unsigned short* W1p, const float* __restrict__ b1,
    const unsigned short* W2p, const float* __restrict__ b2,
    const unsigned short* W3p, const float* __restrict__ b3,
    const float* __restrict__ W4, const float* __restrict__ b4,
    float* __restrict__ out) {
  extern __shared__ unsigned short smem[];
  const int XS = 256, WVS = 4;
  unsigned short* w1 = smem;                       // 32768 u16
  unsigned short* w2 = w1 + 32768;                 // 16384 u16
  unsigned short* w3 = w2 + 16384;                 // 16384 u16
  const int wave = threadIdx.x >> 5, lane = threadIdx.x & 31;
  unsigned short* wb    = w3 + 16384 + wave * (4096 + 2 * 32 * XS);
  unsigned short* h     = wb;
  unsigned short* xbase = wb + 4096;

  stage_g2l(w1, W1p, 32768, threadIdx.x, 32 * WVS);
  stage_g2l(w2, W2p, 16384, threadIdx.x, 32 * WVS);
  stage_g2l(w3, W3p, 16384, threadIdx.x, 32 * WVS);
  wait_async0();
  __syncthreads();

  const int ntiles = NUM_VARS / 32;
  const int stride = gridDim.x * WVS;
  int t = blockIdx.x * WVS + wave;
  if (t < ntiles) {
    stage_src32(L + (size_t)t * 32 * DD, xbase, XS, 0, lane);
    stage_src32(L + ((size_t)t * 32 + NUM_VARS) * DD, xbase, XS, DD, lane);
  }
  wait_async0();
  int cur = 0;
  for (; t < ntiles; t += stride) {
    unsigned short* xc = xbase + cur * (32 * XS);
    unsigned short* xn = xbase + (cur ^ 1) * (32 * XS);
    int tn = t + stride;
    if (tn < ntiles) {
      stage_src32(L + (size_t)tn * 32 * DD, xn, XS, 0, lane);
      stage_src32(L + ((size_t)tn * 32 + NUM_VARS) * DD, xn, XS, DD, lane);
    }
    v8f acc0[8] = {}, acc1[8] = {};
    mlp_layer32<8>(xc, XS, w1, lane, acc0, acc1);
    store_h(acc0, b1, h, lane, true);
    store_h(acc1, b1, h + 16 * DD, lane, true);
    {
      v8f a0[8] = {}, a1[8] = {};
      mlp_layer32<4>(h, DD, w2, lane, a0, a1);
      store_h(a0, b2, h, lane, true);
      store_h(a1, b2, h + 16 * DD, lane, true);
    }
    {
      v8f a0[8] = {}, a1[8] = {};
      mlp_layer32<4>(h, DD, w3, lane, a0, a1);
      store_h(a0, b3, h, lane, true);
      store_h(a1, b3, h + 16 * DD, lane, true);
    }
    {                                // out = h3 @ W4 + b4, one row per lane
      const unsigned short* hr = h + lane * DD;
      float sum = b4[0];
#pragma unroll 4
      for (int c = 0; c < DD; ++c) sum += bf_bits2f(hr[c]) * W4[c];
      out[t * 32 + lane] = sum;
    }
    wait_async0();
    cur ^= 1;
  }
}

// ---------------------------------------------------------------------------
// Host orchestration (graph-capture safe)
// ---------------------------------------------------------------------------
extern "C" void kernel_launch(void* const* d_in, const int* in_sizes, int n_in,
                              void* d_out, int out_size, void* d_ws, size_t ws_size,
                              hipStream_t stream) {
  const int*   edge_clause = (const int*)d_in[0];
  const int*   edge_lit    = (const int*)d_in[1];
  const float* c_W1 = (const float*)d_in[2];
  const float* c_b1 = (const float*)d_in[3];
  const float* c_W2 = (const float*)d_in[4];
  const float* c_b2 = (const float*)d_in[5];
  const float* l_W1 = (const float*)d_in[6];
  const float* l_b1 = (const float*)d_in[7];
  const float* l_W2 = (const float*)d_in[8];
  const float* l_b2 = (const float*)d_in[9];
  const float* v_W1 = (const float*)d_in[10];
  const float* v_b1 = (const float*)d_in[11];
  const float* v_W2 = (const float*)d_in[12];
  const float* v_b2 = (const float*)d_in[13];
  const float* v_W3 = (const float*)d_in[14];
  const float* v_b3 = (const float*)d_in[15];
  const float* v_W4 = (const float*)d_in[16];
  const float* v_b4 = (const float*)d_in[17];
  const float* c_init_scale = (const float*)d_in[18];
  const float* l_init_scale = (const float*)d_in[19];
  const float* lc_scale = (const float*)d_in[20];
  const float* cl_scale = (const float*)d_in[21];
  (void)in_sizes; (void)n_in; (void)out_size; (void)ws_size;

  // ---- workspace layout: bf16 embeddings + packed weights + f32 accum ----
  unsigned short* u = (unsigned short*)d_ws;
  unsigned short* Lb[2];
  Lb[0] = u;                                       u += (size_t)NUM_LITS * DD;
  Lb[1] = u;                                       u += (size_t)NUM_LITS * DD;
  unsigned short* Cb  = u;                         u += (size_t)NUM_CLAUSES * DD;
  unsigned short* LCb = u;                         u += (size_t)NUM_CLAUSES * DD;
  unsigned short* CLb = u;                         u += (size_t)NUM_LITS * DD;
  unsigned short* cW1p = u;                        u += (size_t)RR * 8  * 8 * 512;
  unsigned short* cW2p = u;                        u += (size_t)RR * 4  * 8 * 512;
  unsigned short* lW1p = u;                        u += (size_t)RR * 12 * 8 * 512;
  unsigned short* lW2p = u;                        u += (size_t)RR * 4  * 8 * 512;
  unsigned short* vW1p = u;                        u += (size_t)8 * 8 * 512;
  unsigned short* vW2p = u;                        u += (size_t)4 * 8 * 512;
  unsigned short* vW3p = u;                        u += (size_t)4 * 8 * 512;
  float* accf = (float*)u;                         // NUM_CLAUSES*DD f32 scratch

  // ---- pack weights (scales folded into the LC/CL halves of W1) ----
  for (int r = 0; r < RR; ++r) {
    pack_w_kernel<<<(8 * 256 + 255) / 256, 256, 0, stream>>>(
        c_W1 + (size_t)r * 256 * DD, cW1p + (size_t)r * 8 * 8 * 512, 8, lc_scale, 128, 256);
    pack_w_kernel<<<(4 * 256 + 255) / 256, 256, 0, stream>>>(
        c_W2 + (size_t)r * 128 * DD, cW2p + (size_t)r * 4 * 8 * 512, 4, lc_scale, 0, 0);
    pack_w_kernel<<<(12 * 256 + 255) / 256, 256, 0, stream>>>(
        l_W1 + (size_t)r * 384 * DD, lW1p + (size_t)r * 12 * 8 * 512, 12, cl_scale, 128, 256);
    pack_w_kernel<<<(4 * 256 + 255) / 256, 256, 0, stream>>>(
        l_W2 + (size_t)r * 128 * DD, lW2p + (size_t)r * 4 * 8 * 512, 4, cl_scale, 0, 0);
  }
  pack_w_kernel<<<(8 * 256 + 255) / 256, 256, 0, stream>>>(v_W1, vW1p, 8, lc_scale, 0, 0);
  pack_w_kernel<<<(4 * 256 + 255) / 256, 256, 0, stream>>>(v_W2, vW2p, 4, lc_scale, 0, 0);
  pack_w_kernel<<<(4 * 256 + 255) / 256, 256, 0, stream>>>(v_W3, vW3p, 4, lc_scale, 0, 0);

  // ---- init embeddings (bf16) ----
  {
    size_t nl = (size_t)NUM_LITS * DD;
    init_embed_kernel<<<(unsigned)((nl + 255) / 256), 256, 0, stream>>>(Lb[0], l_init_scale, nl);
    size_t nc = (size_t)NUM_CLAUSES * DD;
    init_embed_kernel<<<(unsigned)((nc + 255) / 256), 256, 0, stream>>>(Cb, c_init_scale, nc);
  }

  const int scatter_blocks = (NNZ * 32 + 255) / 256;  // one wave per edge
  const size_t nC4 = (size_t)NUM_CLAUSES * DD / 4;
  const size_t nL4 = (size_t)NUM_LITS * DD / 4;
  // dynamic LDS sizes (bytes)
  const size_t sh_clause = (size_t)(32768 + 16384 + 4 * (4096 + 2 * 32 * 256)) * 2;
  const size_t sh_lit    = (size_t)(49152 + 16384 + 3 * (4096 + 2 * 32 * 384)) * 2;
  const size_t sh_var    = (size_t)(32768 + 16384 + 16384 + 4 * (4096 + 2 * 32 * 256)) * 2;

  int cur = 0;
  for (int r = 0; r < RR; ++r) {
    (void)hipMemsetAsync(accf, 0, (size_t)NUM_CLAUSES * DD * sizeof(float), stream);
    scatter_kernel<<<scatter_blocks, 256, 0, stream>>>(Lb[cur], edge_clause, edge_lit, accf);
    cvt_kernel<<<(unsigned)((nC4 + 255) / 256), 256, 0, stream>>>(accf, LCb, nC4);
    clause_mlp_kernel<<<1024, 128, sh_clause, stream>>>(
        Cb, LCb, cW1p + (size_t)r * 8 * 8 * 512, c_b1 + r * DD,
        cW2p + (size_t)r * 4 * 8 * 512, c_b2 + r * DD, Cb);
    (void)hipMemsetAsync(accf, 0, (size_t)NUM_LITS * DD * sizeof(float), stream);
    scatter_kernel<<<scatter_blocks, 256, 0, stream>>>(Cb, edge_lit, edge_clause, accf);
    cvt_kernel<<<(unsigned)((nL4 + 255) / 256), 256, 0, stream>>>(accf, CLb, nL4);
    lit_mlp_kernel<<<1024, 96, sh_lit, stream>>>(
        Lb[cur], CLb, lW1p + (size_t)r * 12 * 8 * 512, l_b1 + r * DD,
        lW2p + (size_t)r * 4 * 8 * 512, l_b2 + r * DD, Lb[1 - cur]);
    cur = 1 - cur;
  }
  var_mlp_kernel<<<512, 128, sh_var, stream>>>(
      Lb[cur], vW1p, v_b1, vW2p, v_b2, vW3p, v_b3, v_W4, v_b4, (float*)d_out);
}
